// FrameTransformer_46196668236004
// MI455X (gfx1250) — compile-verified, matrix operand order checked by reading, not run on hardware
//
#include <hip/hip_runtime.h>
#include <cstdint>

typedef unsigned short u16;
typedef unsigned int   u32;
typedef __bf16 bf16;
typedef bf16  bf16x16 __attribute__((ext_vector_type(16)));
typedef float f32x8   __attribute__((ext_vector_type(8)));

#define EPI_RELU_PRE  1
#define EPI_ACC       2
#define EPI_RELU_POST 4
#define GEMM_TRANSB   8

#define BM 128
#define BN 128
#define BK 32
#define SK 40   // padded LDS row stride (bf16 elements) -> 80B, 16B aligned chunks

__device__ __forceinline__ u16 f2bf(float f) {
  u32 x = __float_as_uint(f);
  x += 0x7FFFu + ((x >> 16) & 1u);   // round to nearest even
  return (u16)(x >> 16);
}

__device__ __forceinline__ f32x8 wmma_bf16(bf16x16 a, bf16x16 b, f32x8 c) {
  return __builtin_amdgcn_wmma_f32_16x16x32_bf16(false, a, false, b, (short)0, c, false, false);
}

// async global -> LDS copy of 16 bytes (per lane); tracked by ASYNCcnt
__device__ __forceinline__ void async_b128(const void* gp, void* lp) {
  unsigned lds32 = (unsigned)(uintptr_t)lp;     // LDS aperture: low 32 bits = LDS offset
  unsigned long long ga = (unsigned long long)(uintptr_t)gp;
  asm volatile("global_load_async_to_lds_b128 %0, %1, off"
               :: "v"(lds32), "v"(ga) : "memory");
}
__device__ __forceinline__ void wait_async0() {
  asm volatile("s_wait_asynccnt 0x0" ::: "memory");
}

union Frag { uint4 u[2]; bf16x16 v; };

// ---------------------------------------------------------------------------
// Generic bf16 WMMA GEMM:  C[z] = epi( alpha * A[z] @ B[z]^T + bias )
//   A: M x K (row major, lda), B: N x K (row major, ldb)  [or K x N if TRANSB_T]
//   per-z offsets: zb = z/nH, zh = z%nH ; off = zb*s?b + zh*s?h (elements)
//   SHIFT_T: A rows shifted within 256-row batches (zero-fill OOB) for conv1d
//   Double-buffered LDS: async global->LDS of tile k+1 overlaps WMMA of tile k.
// ---------------------------------------------------------------------------
template<bool TRANSB_T, bool SHIFT_T>
__global__ __launch_bounds__(256) void k_gemm(
    const u16* __restrict__ A, const u16* __restrict__ Bw,
    const float* __restrict__ bias,
    const float* __restrict__ Cin, float* __restrict__ Cout, u16* __restrict__ CoutB,
    int M, int N, int K, int lda, int ldb, int ldc,
    long long sAb, long long sAh, long long sBb, long long sBh,
    long long sCb, long long sCh,
    int nH, int rowShift, float alpha, int flags)
{
  __shared__ u16 As[2][BM * SK];
  __shared__ u16 Bs[2][BN * SK];

  const int z  = blockIdx.z;
  const int zb = z / nH, zh = z % nH;
  const u16* Ap = A  + (long long)zb * sAb + (long long)zh * sAh;
  const u16* Bp = Bw + (long long)zb * sBb + (long long)zh * sBh;
  const long long cOff = (long long)zb * sCb + (long long)zh * sCh;

  const int tid   = threadIdx.x;
  const int lane  = tid & 31;
  const int wave  = tid >> 5;
  const int waveM = wave >> 1;   // 0..3 : 32 rows each
  const int waveN = wave & 1;    // 0..1 : 64 cols each
  const int blockRow = blockIdx.x * BM;
  const int blockCol = blockIdx.y * BN;
  const int half  = lane >> 4;
  const int l15   = lane & 15;
  const int kbase = half * 8;

  f32x8 acc[2][4];
  for (int a = 0; a < 2; a++)
    for (int b = 0; b < 4; b++)
      for (int i = 0; i < 8; i++) acc[a][b][i] = 0.f;

  // stage one K-tile (A 128x32, B 128x32) into LDS buffer `buf`
  auto stage = [&](int kk, int buf) {
#pragma unroll
    for (int p = 0; p < 2; p++) {
      int idx = tid + p * 256;        // 0..511
      int r   = idx >> 2;             // 0..127
      int kc  = (idx & 3) * 8;        // 0,8,16,24
      int gr  = blockRow + r;
      if (SHIFT_T) {
        int w = (gr & 255) + rowShift;
        if (w >= 0 && w < 256) {
          async_b128(Ap + (long long)(gr + rowShift) * lda + kk + kc, &As[buf][r * SK + kc]);
        } else {
          uint4 zz = {0u, 0u, 0u, 0u};
          *(uint4*)&As[buf][r * SK + kc] = zz;
        }
      } else {
        const u16* gp = Ap + (long long)gr * lda + kk + kc;
        async_b128(gp, &As[buf][r * SK + kc]);
        if (p == 0 && kk + BK < K) __builtin_prefetch(gp + BK, 0, 3);
      }
    }
    if (!TRANSB_T) {
#pragma unroll
      for (int p = 0; p < 2; p++) {
        int idx = tid + p * 256;
        int r   = idx >> 2;
        int kc  = (idx & 3) * 8;
        async_b128(Bp + (long long)(blockCol + r) * ldb + kk + kc, &Bs[buf][r * SK + kc]);
      }
    } else {
      // B stored K x N : Bs[n][k] = Bp[(kk+k)*ldb + blockCol + n]  (transpose in LDS)
#pragma unroll
      for (int p = 0; p < 2; p++) {
        int idx = tid + p * 256;
        int kr  = idx >> 4;            // 0..31
        int nc  = (idx & 15) * 8;      // 0..120
        uint4 val = *(const uint4*)(Bp + (long long)(kk + kr) * ldb + blockCol + nc);
        const u16* pv = (const u16*)&val;
#pragma unroll
        for (int j = 0; j < 8; j++) Bs[buf][(nc + j) * SK + kr] = pv[j];
      }
    }
  };

  // prologue: fill buffer 0, wait, barrier
  stage(0, 0);
  wait_async0();
  __syncthreads();

  int cur = 0;
  for (int kk = 0; kk < K; kk += BK) {
    const int nxt = cur ^ 1;
    // prefetch next K-tile into the other buffer (overlaps with WMMA below)
    if (kk + BK < K) stage(kk + BK, nxt);

    Frag af[2];
#pragma unroll
    for (int tm = 0; tm < 2; tm++) {
      int row = waveM * 32 + tm * 16 + l15;
      af[tm].u[0] = *(const uint4*)&As[cur][row * SK + kbase];
      af[tm].u[1] = *(const uint4*)&As[cur][row * SK + kbase + 16];
    }
#pragma unroll
    for (int tn = 0; tn < 4; tn++) {
      int col = waveN * 64 + tn * 16 + l15;
      Frag bf_;
      bf_.u[0] = *(const uint4*)&Bs[cur][col * SK + kbase];
      bf_.u[1] = *(const uint4*)&Bs[cur][col * SK + kbase + 16];
#pragma unroll
      for (int tm = 0; tm < 2; tm++)
        acc[tm][tn] = wmma_bf16(af[tm].v, bf_.v, acc[tm][tn]);
    }

    // next-buffer transfers must land; everyone done reading cur before reuse
    wait_async0();
    __syncthreads();
    cur = nxt;
  }

  // ---- epilogue (C/D layout: VGPR i -> row i + half*8, col l15) ----
#pragma unroll
  for (int tm = 0; tm < 2; tm++) {
#pragma unroll
    for (int tn = 0; tn < 4; tn++) {
      int gcol = blockCol + waveN * 64 + tn * 16 + l15;
      float bv = bias ? bias[gcol] : 0.f;
#pragma unroll
      for (int i = 0; i < 8; i++) {
        int grow = blockRow + waveM * 32 + tm * 16 + half * 8 + i;
        float v = acc[tm][tn][i] * alpha + bv;
        if (flags & EPI_RELU_PRE)  v = fmaxf(v, 0.f);
        long long ci = cOff + (long long)grow * ldc + gcol;
        if (flags & EPI_ACC)       v += Cin[ci];
        if (flags & EPI_RELU_POST) v = fmaxf(v, 0.f);
        if (Cout)  Cout[ci]  = v;
        if (CoutB) CoutB[ci] = f2bf(v);
      }
    }
  }
}

// ---------------------------------------------------------------------------
// fp32 -> bf16 convert (grid stride)
__global__ void k_f2b(const float* __restrict__ s, u16* __restrict__ d, long long n) {
  long long i = (long long)blockIdx.x * blockDim.x + threadIdx.x;
  long long stride = (long long)gridDim.x * blockDim.x;
  for (; i < n; i += stride) d[i] = f2bf(s[i]);
}

// c1R_w (O=1024, I=1024, T=3) -> 3 matrices [t][o][i] bf16
__global__ void k_c1r(const float* __restrict__ s, u16* __restrict__ d) {
  long long n = 3ll * 1024 * 1024;
  long long i = (long long)blockIdx.x * blockDim.x + threadIdx.x;
  long long stride = (long long)gridDim.x * blockDim.x;
  for (; i < n; i += stride) {
    int t  = (int)(i >> 20);
    int o  = (int)(i >> 10) & 1023;
    int ii = (int)i & 1023;
    d[i] = f2bf(s[((long long)(o * 1024 + ii)) * 3 + t]);
  }
}

// er (d=256, c=256) -> erT[c][d] bf16
__global__ void k_ert(const float* __restrict__ s, u16* __restrict__ d) {
  int i = blockIdx.x * 256 + threadIdx.x;   // 65536 total
  int c = i >> 8, dd = i & 255;
  d[i] = f2bf(s[dd * 256 + c]);
}

// bottleneck: channel reduce + BN + relu + (b,bin,w)->(b,w,bin) transpose
__global__ void k_bott(const float* __restrict__ x, const float* __restrict__ bw,
                       const float* __restrict__ g, const float* __restrict__ bb_,
                       const float* __restrict__ m, const float* __restrict__ v,
                       float* __restrict__ outF, u16* __restrict__ outB)
{
  __shared__ float t[16][17];
  int tx = threadIdx.x, ty = threadIdx.y;
  int bin = blockIdx.x * 16 + ty;
  int w   = blockIdx.y * 16 + tx;
  int b   = blockIdx.z;
  float s = 0.f;
#pragma unroll
  for (int c = 0; c < 8; c++)
    s += x[((long long)(b * 8 + c) * 1024 + bin) * 256 + w] * bw[c];
  float inv = rsqrtf(v[0] + 1e-5f);
  float val = fmaxf((s - m[0]) * inv * g[0] + bb_[0], 0.f);
  t[ty][tx] = val;
  __syncthreads();
  int wo = blockIdx.y * 16 + ty;
  int bo = blockIdx.x * 16 + tx;
  long long o = ((long long)b * 256 + wo) * 1024 + bo;
  float ov = t[tx][ty];
  outF[o] = ov;
  outB[o] = f2bf(ov);
}

// xs1 = LN(xs0 + a*sigmoid(gate)) , width 1024
__global__ void k_gluln(const float* __restrict__ xs0, const float* __restrict__ g2,
                        const float* __restrict__ gam, const float* __restrict__ bet,
                        float* __restrict__ outF, u16* __restrict__ outB)
{
  __shared__ float ra[256], rb[256];
  int row = blockIdx.x, tid = threadIdx.x;
  float lv[4];
  float s = 0.f, ss = 0.f;
#pragma unroll
  for (int j = 0; j < 4; j++) {
    int i = tid + j * 256;
    float a  = g2[(long long)row * 2048 + i];
    float gt = g2[(long long)row * 2048 + 1024 + i];
    float val = xs0[(long long)row * 1024 + i] + a * (1.f / (1.f + __expf(-gt)));
    lv[j] = val; s += val; ss += val * val;
  }
  ra[tid] = s; rb[tid] = ss; __syncthreads();
  for (int st = 128; st > 0; st >>= 1) {
    if (tid < st) { ra[tid] += ra[tid + st]; rb[tid] += rb[tid + st]; }
    __syncthreads();
  }
  float mean = ra[0] * (1.f / 1024.f);
  float var  = rb[0] * (1.f / 1024.f) - mean * mean;
  float inv  = rsqrtf(var + 1e-5f);
#pragma unroll
  for (int j = 0; j < 4; j++) {
    int i = tid + j * 256;
    float o = (lv[j] - mean) * inv * gam[i] + bet[i];
    outF[(long long)row * 1024 + i] = o;
    outB[(long long)row * 1024 + i] = f2bf(o);
  }
}

// generic LN with optional partial-width residual; width in {1024,4096}
__global__ void k_ln(const float* __restrict__ in, const float* __restrict__ resid, int residW,
                     const float* __restrict__ gam, const float* __restrict__ bet,
                     float* __restrict__ outF, u16* __restrict__ outB, int width)
{
  __shared__ float ra[256], rb[256];
  int row = blockIdx.x, tid = threadIdx.x;
  int nIt = width >> 8;
  float lv[16];
  float s = 0.f, ss = 0.f;
  for (int j = 0; j < nIt; j++) {
    int i = tid + j * 256;
    float val = in[(long long)row * width + i];
    if (resid && i < residW) val += resid[(long long)row * residW + i];
    lv[j] = val; s += val; ss += val * val;
  }
  ra[tid] = s; rb[tid] = ss; __syncthreads();
  for (int st = 128; st > 0; st >>= 1) {
    if (tid < st) { ra[tid] += ra[tid + st]; rb[tid] += rb[tid + st]; }
    __syncthreads();
  }
  float rw   = 1.f / (float)width;
  float mean = ra[0] * rw;
  float var  = rb[0] * rw - mean * mean;
  float inv  = rsqrtf(var + 1e-5f);
  for (int j = 0; j < nIt; j++) {
    int i = tid + j * 256;
    float o = (lv[j] - mean) * inv * gam[i] + bet[i];
    outF[(long long)row * width + i] = o;
    if (outB) outB[(long long)row * width + i] = f2bf(o);
  }
}

// depthwise conv k=9 pad=4 over w within each batch; h (M x 4096) -> bf16
__global__ void k_dw(const float* __restrict__ h, const float* __restrict__ wdw,
                     u16* __restrict__ out)
{
  int row = blockIdx.x;                       // 0..4095 = b*256 + l
  int c   = blockIdx.y * 256 + threadIdx.x;   // 0..4095
  int b = row >> 8, l = row & 255;
  float s = 0.f;
#pragma unroll
  for (int t = 0; t < 9; t++) {
    int ll = l + t - 4;
    if (ll >= 0 && ll < 256)
      s += h[(long long)(b * 256 + ll) * 4096 + c] * wdw[c * 9 + t];
  }
  out[(long long)row * 4096 + c] = f2bf(s);
}

// softmax with bias: p[b,n,w,k] = softmax_k( qk + |w-k|/2 * dist_w[n,w] + qer[b,n,k,w] )
__global__ void k_smax(const float* __restrict__ sc, const float* __restrict__ qer,
                       const float* __restrict__ dw, u16* __restrict__ pr)
{
  __shared__ float red[256];
  int zz = blockIdx.x;                 // 16384 = b*1024 + n*256 + w
  int w = zz & 255, n = (zz >> 8) & 3, b = zz >> 10;
  int k = threadIdx.x;
  long long rowBase = ((long long)((b * 4 + n) * 256 + w)) * 256;
  float val = sc[rowBase + k]
            + fabsf((float)(w - k)) * 0.5f * dw[n * 256 + w]
            + qer[((long long)((b * 4 + n) * 256 + k)) * 256 + w];
  red[k] = val; __syncthreads();
  for (int st = 128; st > 0; st >>= 1) {
    if (k < st) red[k] = fmaxf(red[k], red[k + st]);
    __syncthreads();
  }
  float mx = red[0]; __syncthreads();
  float e = __expf(val - mx);
  red[k] = e; __syncthreads();
  for (int st = 128; st > 0; st >>= 1) {
    if (k < st) red[k] += red[k + st];
    __syncthreads();
  }
  pr[rowBase + k] = f2bf(e / red[0]);
}

// final (b,w,bin) -> (b,1,bin,w)
__global__ void k_outT(const float* __restrict__ xs, float* __restrict__ out) {
  __shared__ float t[16][17];
  int tx = threadIdx.x, ty = threadIdx.y;
  int b = blockIdx.z;
  int bin = blockIdx.x * 16 + tx;
  int w   = blockIdx.y * 16 + ty;
  t[ty][tx] = xs[((long long)(b * 256 + w)) * 1024 + bin];
  __syncthreads();
  int bino = blockIdx.x * 16 + ty;
  int wo   = blockIdx.y * 16 + tx;
  out[(long long)b * 262144 + (long long)bino * 256 + wo] = t[tx][ty];
}

// ---------------------------------------------------------------------------
static inline void gemm(hipStream_t s, const u16* A, const u16* Bw, const float* bias,
                        const float* Cin, float* Cout, u16* CoutB,
                        int M_, int N, int K, int lda, int ldb, int ldc,
                        long long sAb, long long sAh, long long sBb, long long sBh,
                        long long sCb, long long sCh, int Z, int nH, int rowShift,
                        float alpha, int flags)
{
  dim3 grid(M_ / BM, N / BN, Z);
  if (flags & GEMM_TRANSB)
    k_gemm<true, false><<<grid, 256, 0, s>>>(A, Bw, bias, Cin, Cout, CoutB, M_, N, K,
        lda, ldb, ldc, sAb, sAh, sBb, sBh, sCb, sCh, nH, rowShift, alpha, flags);
  else if (rowShift)
    k_gemm<false, true><<<grid, 256, 0, s>>>(A, Bw, bias, Cin, Cout, CoutB, M_, N, K,
        lda, ldb, ldc, sAb, sAh, sBb, sBh, sCb, sCh, nH, rowShift, alpha, flags);
  else
    k_gemm<false, false><<<grid, 256, 0, s>>>(A, Bw, bias, Cin, Cout, CoutB, M_, N, K,
        lda, ldb, ldc, sAb, sAh, sBb, sBh, sCb, sCh, nH, rowShift, alpha, flags);
}

extern "C" void kernel_launch(void* const* d_in, const int* in_sizes, int n_in,
                              void* d_out, int out_size, void* d_ws, size_t ws_size,
                              hipStream_t stream)
{
  (void)in_sizes; (void)n_in; (void)out_size; (void)ws_size;
  const float* x      = (const float*)d_in[0];
  const float* bott_w = (const float*)d_in[1];
  const float* bn_g   = (const float*)d_in[2];
  const float* bn_b   = (const float*)d_in[3];
  const float* bn_m   = (const float*)d_in[4];
  const float* bn_v   = (const float*)d_in[5];
  const float* glu_w  = (const float*)d_in[6];
  const float* n1g    = (const float*)d_in[7];
  const float* n1b    = (const float*)d_in[8];
  const float* c1L_w  = (const float*)d_in[9];
  const float* c1R_w  = (const float*)d_in[10];
  const float* n2g    = (const float*)d_in[11];
  const float* n2b    = (const float*)d_in[12];
  const float* c2dw_w = (const float*)d_in[13];
  const float* c2pw_w = (const float*)d_in[14];
  const float* n3g    = (const float*)d_in[15];
  const float* n3b    = (const float*)d_in[16];
  const float* qw     = (const float*)d_in[17];
  const float* qb     = (const float*)d_in[18];
  const float* kw     = (const float*)d_in[19];
  const float* kb     = (const float*)d_in[20];
  const float* vw     = (const float*)d_in[21];
  const float* vb     = (const float*)d_in[22];
  const float* ow     = (const float*)d_in[23];
  const float* ob     = (const float*)d_in[24];
  const float* er     = (const float*)d_in[25];
  const float* dist_w = (const float*)d_in[26];
  const float* n4g    = (const float*)d_in[27];
  const float* n4b    = (const float*)d_in[28];
  const float* c3_w   = (const float*)d_in[29];
  const float* c4_w   = (const float*)d_in[30];
  const float* n5g    = (const float*)d_in[31];
  const float* n5b    = (const float*)d_in[32];

  char* ws = (char*)d_ws;
  size_t off = 0;
  auto alloc = [&](size_t bytes) -> char* {
    char* p = ws + off;
    off += (bytes + 255) & ~(size_t)255;
    return p;
  };
  const long long M = 4096;   // B*W

  float* xs0f = (float*)alloc(M * 1024 * 4);
  u16*   xs0b = (u16*)  alloc(M * 1024 * 2);
  char*  big32=         alloc(M * 4096 * 2);   // g(f32 Mx2048) -> tdw(bf16 Mx4096) -> ffh(bf16 Mx4096)
  float* xs1f = (float*)alloc(M * 1024 * 4);
  u16*   xs1b = (u16*)  alloc(M * 1024 * 2);
  char*  hreg =         alloc(M * 4096 * 4);   // h(f32 Mx4096) -> attention buffers
  float* tR   = (float*)alloc(M * 1024 * 4);   // hR accum -> t -> ffo
  float* xs2f = (float*)alloc(M * 1024 * 4);
  u16*   xs2b = (u16*)  alloc(M * 1024 * 2);
  u16*   qbf  = (u16*)  alloc(M * 1024 * 2);
  u16*   kbf  = (u16*)  alloc(M * 1024 * 2);
  u16*   vbf  = (u16*)  alloc(M * 1024 * 2);
  u16* wglu  = (u16*)alloc(2048ll * 1024 * 2);
  u16* wc1L  = (u16*)alloc(4096ll * 1024 * 2);
  u16* wc1r  = (u16*)alloc(3ll * 1024 * 1024 * 2);
  u16* wc2pw = (u16*)alloc(1024ll * 4096 * 2);
  u16* wq    = (u16*)alloc(1024ll * 1024 * 2);
  u16* wk    = (u16*)alloc(1024ll * 1024 * 2);
  u16* wv    = (u16*)alloc(1024ll * 1024 * 2);
  u16* wo_   = (u16*)alloc(1024ll * 1024 * 2);
  u16* werT  = (u16*)alloc(256 * 256 * 2);
  u16* wc3   = (u16*)alloc(4096ll * 1024 * 2);
  u16* wc4   = (u16*)alloc(1024ll * 4096 * 2);

  float* gbuf  = (float*)big32;
  u16*   tdw   = (u16*)big32;
  u16*   ffh   = (u16*)big32;
  float* h     = (float*)hreg;
  float* qerb  = (float*)hreg;                              // after dwconv, h is dead
  float* scr   = (float*)(hreg + 16ll * 1024 * 1024);
  u16*   prb   = (u16*)  (hreg + 32ll * 1024 * 1024);
  u16*   attnb = (u16*)  (hreg + 40ll * 1024 * 1024);
  float* attnp = (float*)(hreg + 48ll * 1024 * 1024);
  float* xs3f  = xs0f;
  u16*   xs3b  = xs0b;
  float* xs4   = xs1f;

  // --- weight conversions to bf16 ---
  k_f2b<<<2048, 256, 0, stream>>>(glu_w,  wglu,  2048ll * 1024);
  k_f2b<<<2048, 256, 0, stream>>>(c1L_w,  wc1L,  4096ll * 1024);
  k_c1r<<<2048, 256, 0, stream>>>(c1R_w,  wc1r);
  k_f2b<<<2048, 256, 0, stream>>>(c2pw_w, wc2pw, 1024ll * 4096);
  k_f2b<<<1024, 256, 0, stream>>>(qw, wq,  1024ll * 1024);
  k_f2b<<<1024, 256, 0, stream>>>(kw, wk,  1024ll * 1024);
  k_f2b<<<1024, 256, 0, stream>>>(vw, wv,  1024ll * 1024);
  k_f2b<<<1024, 256, 0, stream>>>(ow, wo_, 1024ll * 1024);
  k_ert<<<256,  256, 0, stream>>>(er, werT);
  k_f2b<<<2048, 256, 0, stream>>>(c3_w, wc3, 4096ll * 1024);
  k_f2b<<<2048, 256, 0, stream>>>(c4_w, wc4, 1024ll * 4096);

  // --- bottleneck + BN + relu + transpose -> xs0 ---
  k_bott<<<dim3(64, 16, 16), dim3(16, 16), 0, stream>>>(x, bott_w, bn_g, bn_b, bn_m, bn_v,
                                                        xs0f, xs0b);

  // --- GLU: g = xs0 @ glu^T ; xs1 = LN(xs0 + a*sig(b)) ---
  gemm(stream, xs0b, wglu, nullptr, nullptr, gbuf, nullptr,
       4096, 2048, 1024, 1024, 1024, 2048, 0, 0, 0, 0, 0, 0, 1, 1, 0, 1.f, 0);
  k_gluln<<<4096, 256, 0, stream>>>(xs0f, gbuf, n1g, n1b, xs1f, xs1b);

  // --- hL = relu(xs1 @ c1L^T) into h (Mx4096) ---
  gemm(stream, xs1b, wc1L, nullptr, nullptr, h, nullptr,
       4096, 4096, 1024, 1024, 1024, 4096, 0, 0, 0, 0, 0, 0, 1, 1, 0, 1.f, EPI_RELU_PRE);

  // --- hR = relu(conv1d) as 3 shifted GEMMs into tR ---
  gemm(stream, xs1b, wc1r + 0ll * 1048576, nullptr, nullptr, tR, nullptr,
       4096, 1024, 1024, 1024, 1024, 1024, 0, 0, 0, 0, 0, 0, 1, 1, -1, 1.f, 0);
  gemm(stream, xs1b, wc1r + 1ll * 1048576, nullptr, tR, tR, nullptr,
       4096, 1024, 1024, 1024, 1024, 1024, 0, 0, 0, 0, 0, 0, 1, 1, 0, 1.f, EPI_ACC);
  gemm(stream, xs1b, wc1r + 2ll * 1048576, nullptr, tR, tR, nullptr,
       4096, 1024, 1024, 1024, 1024, 1024, 0, 0, 0, 0, 0, 0, 1, 1, 1, 1.f,
       EPI_ACC | EPI_RELU_POST);

  // --- h = LN(hL + pad(hR)) over 4096 (in place, partial residual) ---
  k_ln<<<4096, 256, 0, stream>>>(h, tR, 1024, n2g, n2b, h, nullptr, 4096);

  // --- depthwise conv k=9 -> tdw bf16 ; t = tdw @ c2pw^T ; xs2 = LN(xs1 + t) ---
  k_dw<<<dim3(4096, 16), 256, 0, stream>>>(h, c2dw_w, tdw);
  gemm(stream, tdw, wc2pw, nullptr, nullptr, tR, nullptr,
       4096, 1024, 4096, 4096, 4096, 1024, 0, 0, 0, 0, 0, 0, 1, 1, 0, 1.f, 0);
  k_ln<<<4096, 256, 0, stream>>>(tR, xs1f, 1024, n3g, n3b, xs2f, xs2b, 1024);

  // --- attention: Q,K,V projections (bf16 outputs) ---
  gemm(stream, xs2b, wq, qb, nullptr, nullptr, qbf,
       4096, 1024, 1024, 1024, 1024, 1024, 0, 0, 0, 0, 0, 0, 1, 1, 0, 1.f, 0);
  gemm(stream, xs2b, wk, kb, nullptr, nullptr, kbf,
       4096, 1024, 1024, 1024, 1024, 1024, 0, 0, 0, 0, 0, 0, 1, 1, 0, 1.f, 0);
  gemm(stream, xs2b, wv, vb, nullptr, nullptr, vbf,
       4096, 1024, 1024, 1024, 1024, 1024, 0, 0, 0, 0, 0, 0, 1, 1, 0, 1.f, 0);

  // --- qer[b,n,w,c] = q_head @ er  (batched 64 = 16 b x 4 heads) ---
  gemm(stream, qbf, werT, nullptr, nullptr, qerb, nullptr,
       256, 256, 256, 1024, 256, 256,
       262144, 256, 0, 0, 262144, 65536, 64, 4, 0, 1.f, 0);

  // --- qk = q @ k^T / sqrt(1024) ---
  gemm(stream, qbf, kbf, nullptr, nullptr, scr, nullptr,
       256, 256, 256, 1024, 1024, 256,
       262144, 256, 262144, 256, 262144, 65536, 64, 4, 0, 0.03125f, 0);

  // --- softmax with distance bias + skew(=qer^T) ---
  k_smax<<<16384, 256, 0, stream>>>(scr, qerb, dist_w, prb);

  // --- o = probs @ v  (B is K-major here -> TRANSB), write heads back interleaved ---
  gemm(stream, prb, vbf, nullptr, nullptr, nullptr, attnb,
       256, 256, 256, 256, 1024, 1024,
       262144, 65536, 262144, 256, 262144, 256, 64, 4, 0, 1.f, GEMM_TRANSB);

  // --- out projection ; xs3 = LN(xs2 + attn) ---
  gemm(stream, attnb, wo_, ob, nullptr, attnp, nullptr,
       4096, 1024, 1024, 1024, 1024, 1024, 0, 0, 0, 0, 0, 0, 1, 1, 0, 1.f, 0);
  k_ln<<<4096, 256, 0, stream>>>(attnp, xs2f, 1024, n4g, n4b, xs3f, xs3b, 1024);

  // --- FFN ; xs4 = LN(xs3 + ffn) ---
  gemm(stream, xs3b, wc3, nullptr, nullptr, nullptr, ffh,
       4096, 4096, 1024, 1024, 1024, 4096, 0, 0, 0, 0, 0, 0, 1, 1, 0, 1.f, EPI_RELU_PRE);
  gemm(stream, ffh, wc4, nullptr, nullptr, tR, nullptr,
       4096, 1024, 4096, 4096, 4096, 1024, 0, 0, 0, 0, 0, 0, 1, 1, 0, 1.f, 0);
  k_ln<<<4096, 256, 0, stream>>>(tR, xs3f, 1024, n5g, n5b, xs4, nullptr, 1024);

  // --- final transpose to (B,1,BINS,W) ---
  k_outT<<<dim3(64, 16, 16), dim3(16, 16), 0, stream>>>(xs4, (float*)d_out);
}
